// SimilarityModel_1022202216874
// MI455X (gfx1250) — matrix operand order repaired and checked by reading, not compiled, not run on hardware
//
#include <hip/hip_runtime.h>
#include <hip/hip_bf16.h>

typedef __attribute__((ext_vector_type(16))) _Float16 v16h;
typedef __attribute__((ext_vector_type(8)))  _Float16 v8h;
typedef __attribute__((ext_vector_type(8)))  float    v8f;

typedef unsigned int u32x4 __attribute__((ext_vector_type(4)));
typedef int          i32x4 __attribute__((ext_vector_type(4)));
typedef int          i32x8 __attribute__((ext_vector_type(8)));

#define DDIM   512
#define N1     4096
#define N2     49152
#define SPLITN 24576
#define CAP    65536
#define SIM_T  0.2f
#define AMB_T  0.17f

#define BLK_M  32          // A rows staged in LDS per block
#define WAVE_N 32          // N columns per wave
#define WAVES  8           // waves per block
#define BLK_N  (WAVE_N * WAVES)

// ---------------------------------------------------------------------------
// Normalize each 512-elem row to unit L2 norm and downconvert to f16.
// ---------------------------------------------------------------------------
__global__ __launch_bounds__(256) void normalize_rows_f16(
    const float* __restrict__ src, _Float16* __restrict__ dst, int rows) {
  int row = blockIdx.x;
  if (row >= rows) return;
  const float* s = src + (size_t)row * DDIM;
  float v0 = s[threadIdx.x];
  float v1 = s[threadIdx.x + 256];
  __shared__ float red[256];
  red[threadIdx.x] = v0 * v0 + v1 * v1;
  __syncthreads();
  for (int off = 128; off > 0; off >>= 1) {
    if (threadIdx.x < off) red[threadIdx.x] += red[threadIdx.x + off];
    __syncthreads();
  }
  float inv = rsqrtf(fmaxf(red[0], 1e-24f));
  _Float16* d = dst + (size_t)row * DDIM;
  d[threadIdx.x]       = (_Float16)(v0 * inv);
  d[threadIdx.x + 256] = (_Float16)(v1 * inv);
}

// ---------------------------------------------------------------------------
// Initialize output pair buffers to -1 and zero the 4 compaction counters.
// ---------------------------------------------------------------------------
__global__ void init_out(int* __restrict__ out, int* __restrict__ counters, int n) {
  int gid = blockIdx.x * blockDim.x + threadIdx.x;
  if (gid < n) out[gid] = -1;
  if (gid < 4) counters[gid] = 0;
}

// ---------------------------------------------------------------------------
// TDM descriptor build + issue: stage BLK_M x DDIM f16 tile, global -> LDS.
// D# layout per CDNA5 ISA ch.8 (Tensor DMA Descriptor).
// ---------------------------------------------------------------------------
#if __has_builtin(__builtin_amdgcn_tensor_load_to_lds)
__device__ __forceinline__ void tdm_issue_a_tile(const _Float16* gsrc,
                                                 void* lds_dst) {
  unsigned long long ga = (unsigned long long)(size_t)gsrc;      // byte addr
  unsigned           la = (unsigned)(size_t)lds_dst;             // LDS offset

  // Group 0: [1:0]=count=1 | [63:32]=lds_addr | [120:64]=global_addr | type=2
  u32x4 g0;
  g0[0] = 1u;                                    // count=1, no gather
  g0[1] = la;                                    // lds_addr
  g0[2] = (unsigned)(ga & 0xFFFFFFFFu);          // global_addr[31:0]
  g0[3] = (unsigned)((ga >> 32) & 0x01FFFFFFu)   // global_addr[56:32]
        | (2u << 30);                            // type = 2 ("image")
  // Group 1: data_size=1 (2 bytes), dims/strides
  i32x8 g1;
  g1[0] = (1 << 16);                             // wg_mask=0, data_size=2B
  g1[1] = (int)((DDIM & 0xFFFF) << 16);          // tensor_dim0[15:0] @ dw1[31:16]
  g1[2] = (int)(((DDIM >> 16) & 0xFFFF)          // tensor_dim0[31:16]
        | ((N1 & 0xFFFF) << 16));                // tensor_dim1[15:0]
  g1[3] = (int)(((N1 >> 16) & 0xFFFF)            // tensor_dim1[31:16]
        | ((DDIM & 0xFFFF) << 16));              // tile_dim0 = DDIM
  g1[4] = (int)(BLK_M & 0xFFFF);                 // tile_dim1 = BLK_M, tile_dim2=0
  g1[5] = (int)DDIM;                             // tensor_dim0_stride low32
  g1[6] = 0;
  g1[7] = 0;
  i32x4 gz = {0, 0, 0, 0};                       // groups 2/3 unused (2D tile)
#if __clang_major__ >= 23
  i32x8 gz8 = {0, 0, 0, 0, 0, 0, 0, 0};
  __builtin_amdgcn_tensor_load_to_lds(g0, g1, gz, gz, gz8, 0);
#else
  __builtin_amdgcn_tensor_load_to_lds(g0, g1, gz, gz, 0);
#endif
  __builtin_amdgcn_s_wait_tensorcnt(0);
}
#endif

// ---------------------------------------------------------------------------
// Fused WMMA GEMM (f16 in, f32 acc) + threshold + compaction epilogue.
// Wave tile: 32(M) x 32(N): 2 A frags (LDS) x 2 B frags, 4 accumulators.
// Block: 8 waves share a 32-row A tile staged once via the Tensor Data Mover.
// Epilogue guarded by a wave-wide ballot early-out (hits are ~4-sigma rare).
// ---------------------------------------------------------------------------
__global__ __launch_bounds__(256) void sim_gemm_compact(
    const _Float16* __restrict__ A,        // [N1, DDIM] normalized f1
    const _Float16* __restrict__ B,        // [N2, DDIM] normalized f2
    const int* __restrict__ labels,  const int* __restrict__ counts,
    const int* __restrict__ indexes,
    const int* __restrict__ labels_db, const int* __restrict__ counts_db,
    const int* __restrict__ indexes_db,
    int* __restrict__ out,                 // [4*CAP*2] ints (sim then amb)
    int* __restrict__ counters)            // [4]: sim s0, sim s1, amb s0, amb s1
{
  __shared__ _Float16 aTile[BLK_M * DDIM];           // 32 KB

  const int lane    = threadIdx.x & 31;
  const int wave    = threadIdx.x >> 5;
  const int mBase   = blockIdx.y * BLK_M;
  const int nBase   = blockIdx.x * BLK_N + wave * WAVE_N;
  const int halfSel = lane >> 4;                     // 0: lanes 0-15, 1: 16-31
  const int l15     = lane & 15;

  // Stage A tile once per block.
#if __has_builtin(__builtin_amdgcn_tensor_load_to_lds)
  if (wave == 0) tdm_issue_a_tile(A + (size_t)mBase * DDIM, aTile);
#else
  {  // cooperative fallback: all 256 threads copy
    v8h* d = (v8h*)aTile;
    const v8h* s = (const v8h*)(A + (size_t)mBase * DDIM);
    #pragma unroll
    for (int i = 0; i < (BLK_M * DDIM / 8) / 256; ++i)
      d[threadIdx.x + i * 256] = s[threadIdx.x + i * 256];
  }
#endif
  __syncthreads();

  // LDS read pointers for the two A sub-tiles (rows l15 and l15+16).
  const _Float16* aRow0 = aTile + (size_t)l15        * DDIM + halfSel * 16;
  const _Float16* aRow1 = aTile + (size_t)(l15 + 16) * DDIM + halfSel * 16;
  // Global B: lane holds column n; 16 contiguous K per load.
  const _Float16* bRow0 = B + (size_t)(nBase +      l15) * DDIM + halfSel * 16;
  const _Float16* bRow1 = B + (size_t)(nBase + 16 + l15) * DDIM + halfSel * 16;

  v8f c00 = {}, c01 = {}, c10 = {}, c11 = {};

  #pragma unroll 2
  for (int kk = 0; kk < DDIM; kk += 32) {
    v8h a0lo = *(const v8h*)(aRow0 + kk);            // ds_load_b128
    v8h a0hi = *(const v8h*)(aRow0 + kk + 8);
    v8h a1lo = *(const v8h*)(aRow1 + kk);
    v8h a1hi = *(const v8h*)(aRow1 + kk + 8);
    v16h a0 = __builtin_shufflevector(a0lo, a0hi,
                0,1,2,3,4,5,6,7,8,9,10,11,12,13,14,15);
    v16h a1 = __builtin_shufflevector(a1lo, a1hi,
                0,1,2,3,4,5,6,7,8,9,10,11,12,13,14,15);
    v16h b0 = *(const v16h*)(bRow0 + kk);
    v16h b1 = *(const v16h*)(bRow1 + kk);
    if (kk + 64 < DDIM) {
      __builtin_prefetch(bRow0 + kk + 64, 0, 1);     // global_prefetch_b8
      __builtin_prefetch(bRow1 + kk + 64, 0, 1);
    }
    c00 = __builtin_amdgcn_wmma_f32_16x16x32_f16(false, a0, false, b0,
                                                 (short)0, c00, false, false);
    c01 = __builtin_amdgcn_wmma_f32_16x16x32_f16(false, a0, false, b1,
                                                 (short)0, c01, false, false);
    c10 = __builtin_amdgcn_wmma_f32_16x16x32_f16(false, a1, false, b0,
                                                 (short)0, c10, false, false);
    c11 = __builtin_amdgcn_wmma_f32_16x16x32_f16(false, a1, false, b1,
                                                 (short)0, c11, false, false);
  }

  // ---- Wave-uniform early-out: any sim above the lower threshold? ----
  float mx = -1.0f;
  #pragma unroll
  for (int r = 0; r < 8; ++r) {
    mx = fmaxf(mx, fmaxf(fmaxf(c00[r], c01[r]), fmaxf(c10[r], c11[r])));
  }
  if (__builtin_amdgcn_ballot_w32(mx > AMB_T) == 0u) return;

  // ---- Rare path: full masked threshold + compaction ----
  // db-side values are fixed per lane: columns gn0 = nBase+l15, gn1 = +16.
  const int gn0 = nBase + l15, gn1 = gn0 + 16;
  const int ln_[2]  = { labels_db[gn0],  labels_db[gn1]  };
  const int in_[2]  = { indexes_db[gn0], indexes_db[gn1] };
  const int cn_[2]  = { counts_db[gn0],  counts_db[gn1]  };
  const int s_[2]   = { gn0 / SPLITN,    gn1 / SPLITN    };

  #pragma unroll
  for (int mi = 0; mi < 2; ++mi) {
    const int mOff = mBase + mi * 16 + 8 * halfSel;
    #pragma unroll
    for (int r = 0; r < 8; ++r) {
      const int m  = mOff + r;
      const int lm = labels[m];
      const int cm = counts[m];
      const int im = indexes[m];
      #pragma unroll
      for (int t = 0; t < 2; ++t) {
        const float sim = mi ? (t ? c11[r] : c10[r])
                             : (t ? c01[r] : c00[r]);
        if (sim <= AMB_T) continue;
        if (im >= in_[t] || lm == ln_[t]) continue;       // validity mask
        const int s = s_[t];
        if (sim > SIM_T) {
          int pos = atomicAdd(&counters[s], 1);
          if (pos < CAP) {
            int* p = out + ((size_t)s * CAP + pos) * 2;
            p[0] = lm; p[1] = ln_[t];
          }
        } else {
          int pos = atomicAdd(&counters[2 + s], 1);
          if (pos < CAP) {
            int* p = out + (size_t)2 * CAP * 2 + ((size_t)s * CAP + pos) * 2;
            const bool swap = cm > cn_[t];
            p[0] = swap ? ln_[t] : lm;
            p[1] = swap ? lm : ln_[t];
          }
        }
      }
    }
  }
}

// ---------------------------------------------------------------------------
extern "C" void kernel_launch(void* const* d_in, const int* in_sizes, int n_in,
                              void* d_out, int out_size, void* d_ws, size_t ws_size,
                              hipStream_t stream) {
  (void)in_sizes; (void)n_in; (void)out_size; (void)ws_size;

  const float* features    = (const float*)d_in[0];
  const int*   labels      = (const int*)  d_in[1];
  const int*   counts      = (const int*)  d_in[2];
  const int*   indexes     = (const int*)  d_in[3];
  const float* features_db = (const float*)d_in[4];
  const int*   labels_db   = (const int*)  d_in[5];
  const int*   counts_db   = (const int*)  d_in[6];
  const int*   indexes_db  = (const int*)  d_in[7];

  // Workspace: [0,16) counters | [256,..) f16 f1 (4 MB) then f16 f2 (48 MB)
  int*      counters = (int*)d_ws;
  _Float16* f1h      = (_Float16*)((char*)d_ws + 256);
  _Float16* f2h      = f1h + (size_t)N1 * DDIM;

  int* out      = (int*)d_out;
  int  totalOut = 4 * CAP * 2;   // 524288 ints

  init_out<<<(totalOut + 255) / 256, 256, 0, stream>>>(out, counters, totalOut);
  normalize_rows_f16<<<N1, 256, 0, stream>>>(features,    f1h, N1);
  normalize_rows_f16<<<N2, 256, 0, stream>>>(features_db, f2h, N2);

  dim3 grid(N2 / BLK_N, N1 / BLK_M);   // (192, 128) blocks, 8 waves each
  sim_gemm_compact<<<grid, 256, 0, stream>>>(f1h, f2h,
                                             labels, counts, indexes,
                                             labels_db, counts_db, indexes_db,
                                             out, counters);
}
